// PhysicsResidual_18459769438382
// MI455X (gfx1250) — compile-verified, matrix operand order checked by reading, not run on hardware
//
#include <hip/hip_runtime.h>
#include <cmath>

// Wave-equation PINN residual via forward-mode second-order (Taylor-jet) AD.
// Per point: 9 channels (primal, 4 first-derivs, 4 second-derivs) of width 128.
// Each hidden layer = one GEMM  Z = W(128x128) @ X(128 x 144)  done with
// v_wmma_f32_16x16x32_f16, followed by the tanh jet combine:
//   a = tanh(z); g = 1-a^2; a1 = g*z1; a2 = g*z2 - 2*a*g*z1^2
// Head (1x128, linear): out = W6 . (h2_t - h2_x - h2_y - h2_z); b6 irrelevant.
// W1..W5 are pre-converted to f16 once (prologue kernel -> d_ws) so the main
// kernel's A fragments are straight b128 loads with no conversion VALU.

typedef _Float16 v16h __attribute__((ext_vector_type(16)));
typedef _Float16 v8h  __attribute__((ext_vector_type(8)));
typedef _Float16 v4h  __attribute__((ext_vector_type(4)));
typedef _Float16 v2h  __attribute__((ext_vector_type(2)));
typedef float    v8f  __attribute__((ext_vector_type(8)));

union U16 { v8h h[2]; v16h v; };
union U8  { v2h p[4]; v8h  v; };

#define CS   136   // LDS column stride in halves (272B -> 16B aligned columns)
#define NCOL 144   // 9 channels * 16 points
#define PTS  16    // points per workgroup

// tanh via the transcendental pipe: 1 - 2/(exp2(2*log2(e)*x)+1).
// v_exp_f32 + v_rcp_f32 + 3 VALU ops; saturates correctly at +/-inf.
__device__ __forceinline__ float fast_tanh(float x) {
    const float e = __builtin_amdgcn_exp2f(x * 2.88539008177792681472f);
    return 1.0f - 2.0f * __builtin_amdgcn_rcpf(e + 1.0f);
}

// ---- Prologue: convert W1..W5 (5 x 128x128 f32) to f16 in d_ws ------------
__global__ __launch_bounds__(256)
void convert_weights_kernel(const float* __restrict__ W1,
                            const float* __restrict__ W2,
                            const float* __restrict__ W3,
                            const float* __restrict__ W4,
                            const float* __restrict__ W5,
                            _Float16* __restrict__ Wh)
{
    const int t = blockIdx.x * 256 + threadIdx.x;   // 0..20479 (x4 floats)
    const float* const Ws[5] = {W1, W2, W3, W4, W5};
    const int l   = t >> 12;            // 4096 float4 per 128x128 layer
    const int off = (t & 4095) * 4;
    const float4 q = *(const float4*)(Ws[l] + off);
    v4h o;
    o[0] = (_Float16)q.x; o[1] = (_Float16)q.y;
    o[2] = (_Float16)q.z; o[3] = (_Float16)q.w;
    *(v4h*)(Wh + l * 16384 + off) = o;
}

template<bool HALFW>
__global__ __launch_bounds__(256)
void wave_pinn_residual_kernel(const float* __restrict__ tt,
                               const float* __restrict__ xx,
                               const float* __restrict__ yy,
                               const float* __restrict__ zz,
                               const float* __restrict__ W0,
                               const float* __restrict__ b0,
                               const float* __restrict__ W1,
                               const float* __restrict__ b1,
                               const float* __restrict__ W2,
                               const float* __restrict__ b2,
                               const float* __restrict__ W3,
                               const float* __restrict__ b3,
                               const float* __restrict__ W4,
                               const float* __restrict__ b4,
                               const float* __restrict__ W5,
                               const float* __restrict__ b5,
                               const float* __restrict__ W6,
                               const _Float16* __restrict__ Wh,   // f16 W1..W5 (or null)
                               float* __restrict__ out)
{
    __shared__ __align__(16) _Float16 Xs[NCOL * CS];  // column-major: Xs[col*CS + row]
    __shared__ float red[256];

    const int tid  = threadIdx.x;
    const int lane = tid & 31;
    const int wv   = tid >> 5;       // wave id 0..7 -> output row block
    const int m    = lane & 15;      // M (A rows) and N (B/D cols) index
    const int hi   = lane >> 4;      // half-wave selector (K / row-half split)
    const int row0 = wv * 16;
    const int g0   = blockIdx.x * PTS;

    // ---------------- Layer 0: (128x4) input layer, seed the 9 channels ----
    {
        const int p  = tid >> 4;          // point 0..15
        const int nb = (tid & 15) * 8;    // neuron block base
        const int gp = g0 + p;
        const float pt0 = tt[gp], pt1 = xx[gp], pt2 = yy[gp], pt3 = zz[gp];
        v8h c0h, c1h[4], c2h[4];
        #pragma unroll
        for (int j = 0; j < 8; ++j) {
            const int n = nb + j;
            const float w0 = W0[n * 4 + 0], w1 = W0[n * 4 + 1];
            const float w2 = W0[n * 4 + 2], w3 = W0[n * 4 + 3];
            const float zv = b0[n] + w0 * pt0 + w1 * pt1 + w2 * pt2 + w3 * pt3;
            const float a  = fast_tanh(zv);
            const float g  = 1.0f - a * a;
            const float m2ag = -2.0f * a * g;
            const float ws[4] = {w0, w1, w2, w3};
            c0h[j] = (_Float16)a;
            #pragma unroll
            for (int i = 0; i < 4; ++i) {
                c1h[i][j] = (_Float16)(g * ws[i]);
                c2h[i][j] = (_Float16)(m2ag * ws[i] * ws[i]);
            }
        }
        *(v8h*)(&Xs[(0 * PTS + p) * CS + nb]) = c0h;
        #pragma unroll
        for (int i = 0; i < 4; ++i) {
            *(v8h*)(&Xs[((1 + i) * PTS + p) * CS + nb]) = c1h[i];
            *(v8h*)(&Xs[((5 + i) * PTS + p) * CS + nb]) = c2h[i];
        }
    }
    __syncthreads();

    // ---------------- Hidden layers 1..5: WMMA GEMM + tanh jet combine -----
    const float* const Wl[5] = {W1, W2, W3, W4, W5};
    const float* const bl[5] = {b1, b2, b3, b4, b5};

    #pragma unroll
    for (int l = 0; l < 5; ++l) {
        // A fragments: rows [row0, row0+16) of W, K = 0..127 in 4 tiles of 32.
        // 16-bit A 16x32 lane layout: lanes 0-15 hold K 0..7 & 16..23,
        // lanes 16-31 hold K 8..15 & 24..31 (8-half contiguous segments).
        v16h afr[4];
        if constexpr (HALFW) {
            const _Float16* const wb = Wh + l * 16384 + (row0 + m) * 128;
            #pragma unroll
            for (int kt = 0; kt < 4; ++kt) {
                const int o1 = kt * 32 + hi * 8;
                U16 u;
                u.h[0] = *(const v8h*)(wb + o1);        // global_load_b128
                u.h[1] = *(const v8h*)(wb + o1 + 16);   // global_load_b128
                afr[kt] = u.v;
            }
        } else {
            const float* const wbase = Wl[l] + (row0 + m) * 128;
            #pragma unroll
            for (int kt = 0; kt < 4; ++kt) {
                const int o1 = kt * 32 + hi * 8;
                const float4 q0 = *(const float4*)(wbase + o1);
                const float4 q1 = *(const float4*)(wbase + o1 + 4);
                const float4 q2 = *(const float4*)(wbase + o1 + 16);
                const float4 q3 = *(const float4*)(wbase + o1 + 20);
                v16h a;
                a[0]  = (_Float16)q0.x; a[1]  = (_Float16)q0.y;
                a[2]  = (_Float16)q0.z; a[3]  = (_Float16)q0.w;
                a[4]  = (_Float16)q1.x; a[5]  = (_Float16)q1.y;
                a[6]  = (_Float16)q1.z; a[7]  = (_Float16)q1.w;
                a[8]  = (_Float16)q2.x; a[9]  = (_Float16)q2.y;
                a[10] = (_Float16)q2.z; a[11] = (_Float16)q2.w;
                a[12] = (_Float16)q3.x; a[13] = (_Float16)q3.y;
                a[14] = (_Float16)q3.z; a[15] = (_Float16)q3.w;
                afr[kt] = a;
            }
        }

        float biasv[8];
        #pragma unroll
        for (int v = 0; v < 8; ++v) biasv[v] = bl[l][row0 + v + hi * 8];

        // K-outer / tile-inner: batch the 18 ds_load_b128 for one K-step,
        // then drain 9 independent WMMAs.
        v8f acc[9];
        #pragma unroll
        for (int c = 0; c < 9; ++c)
            acc[c] = (v8f){0.f, 0.f, 0.f, 0.f, 0.f, 0.f, 0.f, 0.f};

        #pragma unroll
        for (int kt = 0; kt < 4; ++kt) {
            const int o1 = kt * 32 + hi * 8;
            v16h bfs[9];
            #pragma unroll
            for (int c = 0; c < 9; ++c) {
                const _Float16* const colp = &Xs[(c * 16 + m) * CS];
                U16 u;
                u.h[0] = *(const v8h*)(colp + o1);        // ds_load_b128
                u.h[1] = *(const v8h*)(colp + o1 + 16);   // ds_load_b128
                bfs[c] = u.v;
            }
            #pragma unroll
            for (int c = 0; c < 9; ++c) {
                acc[c] = __builtin_amdgcn_wmma_f32_16x16x32_f16(
                            false, afr[kt], false, bfs[c], (short)0, acc[c], false, false);
            }
        }
        __syncthreads();   // all waves done reading Xs -> safe to overwrite

        // tanh jet combine, write next activations in place (f16).
        // Lane's 8 D slots are 8 consecutive rows of column m -> b128 stores.
        // Packed-f16 derivative math with union-based packing (no inserts):
        //   ap = packed tanh(z);  gp = 1 - ap*ap;  mp = -2*(ap*gp)
        //   h1 = gp*z1;  h2 = gp*z2 + mp*(z1*z1)
        const int rbase = row0 + hi * 8;
        const v2h one  = {(_Float16)1.0f,  (_Float16)1.0f};
        const v2h mtwo = {(_Float16)-2.0f, (_Float16)-2.0f};
        U8 phu;
        v2h ap[4], gpk[4], mpk[4];
        #pragma unroll
        for (int k = 0; k < 4; ++k) {
            const float a0 = fast_tanh(acc[0][2 * k]     + biasv[2 * k]);
            const float a1 = fast_tanh(acc[0][2 * k + 1] + biasv[2 * k + 1]);
            v2h p;
            p[0] = (_Float16)a0; p[1] = (_Float16)a1;    // v_cvt_pk_f16_f32
            ap[k] = p; phu.p[k] = p;
            gpk[k] = one - p * p;                        // v_pk_fma_f16 (neg mod)
            mpk[k] = mtwo * (p * gpk[k]);                // 2x v_pk_mul_f16
        }
        *(v8h*)(&Xs[(0 * 16 + m) * CS + rbase]) = phu.v;
        #pragma unroll
        for (int i = 0; i < 4; ++i) {
            U8 h1u, h2u;
            #pragma unroll
            for (int k = 0; k < 4; ++k) {
                v2h z1p, z2p;
                z1p[0] = (_Float16)acc[1 + i][2 * k];
                z1p[1] = (_Float16)acc[1 + i][2 * k + 1];
                z2p[0] = (_Float16)acc[5 + i][2 * k];
                z2p[1] = (_Float16)acc[5 + i][2 * k + 1];
                h1u.p[k] = gpk[k] * z1p;                          // v_pk_mul_f16
                h2u.p[k] = gpk[k] * z2p + mpk[k] * (z1p * z1p);   // v_pk_mul/fma
            }
            *(v8h*)(&Xs[((1 + i) * 16 + m) * CS + rbase]) = h1u.v;
            *(v8h*)(&Xs[((5 + i) * 16 + m) * CS + rbase]) = h2u.v;
        }
        __syncthreads();
    }

    // ---------------- Head: out = W6 . (h2_t - h2_x - h2_y - h2_z) ---------
    {
        const int p  = tid >> 4;          // point 0..15
        const int nb = (tid & 15) * 8;    // neuron block base
        const v8h x5 = *(const v8h*)(&Xs[(5 * 16 + p) * CS + nb]);
        const v8h x6 = *(const v8h*)(&Xs[(6 * 16 + p) * CS + nb]);
        const v8h x7 = *(const v8h*)(&Xs[(7 * 16 + p) * CS + nb]);
        const v8h x8 = *(const v8h*)(&Xs[(8 * 16 + p) * CS + nb]);
        const float4 w6a = *(const float4*)(W6 + nb);
        const float4 w6b = *(const float4*)(W6 + nb + 4);
        const float w6[8] = {w6a.x, w6a.y, w6a.z, w6a.w, w6b.x, w6b.y, w6b.z, w6b.w};
        float partial = 0.0f;
        #pragma unroll
        for (int j = 0; j < 8; ++j) {
            const float c2 = (float)x5[j] - (float)x6[j] - (float)x7[j] - (float)x8[j];
            partial += w6[j] * c2;
        }
        red[tid] = partial;
        __syncthreads();
        if (tid < PTS) {
            float s = 0.0f;
            #pragma unroll
            for (int k = 0; k < 16; ++k) s += red[tid * 16 + k];
            out[g0 + tid] = s;
        }
    }
}

extern "C" void kernel_launch(void* const* d_in, const int* in_sizes, int n_in,
                              void* d_out, int out_size, void* d_ws, size_t ws_size,
                              hipStream_t stream) {
    // setup_inputs() order: t, x, y, z, then (W0,b0)...(W6,b6)
    const float* t  = (const float*)d_in[0];
    const float* x  = (const float*)d_in[1];
    const float* y  = (const float*)d_in[2];
    const float* z  = (const float*)d_in[3];
    const float* W0 = (const float*)d_in[4];
    const float* b0 = (const float*)d_in[5];
    const float* W1 = (const float*)d_in[6];
    const float* b1 = (const float*)d_in[7];
    const float* W2 = (const float*)d_in[8];
    const float* b2 = (const float*)d_in[9];
    const float* W3 = (const float*)d_in[10];
    const float* b3 = (const float*)d_in[11];
    const float* W4 = (const float*)d_in[12];
    const float* b4 = (const float*)d_in[13];
    const float* W5 = (const float*)d_in[14];
    const float* b5 = (const float*)d_in[15];
    const float* W6 = (const float*)d_in[16];
    // b6 (d_in[17]) unused: head is linear, vanishes in the 2nd-deriv residual.
    float* out = (float*)d_out;

    const int n = in_sizes[0];          // 65536 -> divisible by 16 (no tail)
    const int nblocks = n / PTS;
    const size_t need = (size_t)5 * 16384 * sizeof(_Float16);   // 160 KB

    if (ws_size >= need) {
        _Float16* Wh = (_Float16*)d_ws;
        convert_weights_kernel<<<80, 256, 0, stream>>>(W1, W2, W3, W4, W5, Wh);
        wave_pinn_residual_kernel<true><<<nblocks, 256, 0, stream>>>(
            t, x, y, z, W0, b0, W1, b1, W2, b2, W3, b3, W4, b4, W5, b5, W6,
            Wh, out);
    } else {
        wave_pinn_residual_kernel<false><<<nblocks, 256, 0, stream>>>(
            t, x, y, z, W0, b0, W1, b1, W2, b2, W3, b3, W4, b4, W5, b5, W6,
            nullptr, out);
    }
}
